// PointMamba_56951266345004
// MI455X (gfx1250) — compile-verified, matrix operand order checked by reading, not run on hardware
//
#include <hip/hip_runtime.h>
#include <math.h>

// ---------------- model constants ----------------
static const int BB  = 4;     // batch
static const int NN  = 1024;  // points
static const int GG  = 128;   // groups
static const int KK  = 32;    // knn
static const int DM  = 384;   // d_model
static const int LL  = 3 * GG;// sequence length = 384
static const int DI  = 768;   // d_inner
static const int DS  = 16;    // d_state
static const int DTR = 24;    // dt_rank
static const int DEPTH = 12;
static const int CLS = 40;

#define BN_SCALE_C 0.9999950000374997f  // 1/sqrt(1+1e-5)

typedef __attribute__((ext_vector_type(16))) _Float16 v16h;
typedef __attribute__((ext_vector_type(8)))  _Float16 v8h;
typedef __attribute__((ext_vector_type(8)))  float    v8f;

union HFrag { v16h v; v8h q[2]; };
union AFrag { v8f  v; float f[8]; };

// ------------- f32 -> padded f16 staging -------------
// dst is [Mp x Kp] halves, zero-filled outside [M x Kd]; src row stride = lda.
__global__ void cvt_pad_kernel(const float* __restrict__ A, int M, int Kd, int lda,
                               _Float16* __restrict__ dst, int Mp, int Kp) {
  int idx = blockIdx.x * blockDim.x + threadIdx.x;
  if (idx >= Mp * Kp) return;
  int r = idx / Kp, k = idx - r * Kp;
  float v = (r < M && k < Kd) ? A[(size_t)r * lda + k] : 0.0f;
  dst[idx] = (_Float16)v;
}

// ------------- WMMA GEMM on padded f16: C = epi(Ah[Mp,Kp] @ Wh[Np,Kp]^T + bias) -------------
// All fragment loads are unconditional 16B vector loads (operands are zero-padded).
// A-fragment (16x32 f16, wave32): lane l holds row (l&15); halves 0..7 = K k0+g..k0+g+7,
// halves 8..15 = K k0+16+g.., g = (l>=16)?8:0. B column n == Wh row n.
// EPI: 0 = bias(optional), 1 = bias+BN+ReLU, 2 = bias+GELU(exact), 3 = bias+softplus
template <int EPI>
__global__ void __launch_bounds__(32)
gemm_wmma_kernel(const _Float16* __restrict__ Ah,
                 const _Float16* __restrict__ Wh,
                 const float* __restrict__ bias,
                 const float* __restrict__ gamma,
                 const float* __restrict__ beta,
                 float* __restrict__ C,        // optional f32 out, row stride N
                 _Float16* __restrict__ Ch,    // optional f16 out, row stride N
                 int M, int N, int Kp, int tilesN) {
  const int tile = blockIdx.x;
  const int tm = (tile / tilesN) * 16;
  const int tn = (tile % tilesN) * 16;
  const int lane = threadIdx.x & 31;
  const int l15  = lane & 15;
  const int grp  = (lane >> 4) << 3;   // 0 or 8
  const _Float16* arow = Ah + (size_t)(tm + l15) * Kp;
  const _Float16* wrow = Wh + (size_t)(tn + l15) * Kp;

  AFrag acc; acc.v = (v8f){0.f,0.f,0.f,0.f,0.f,0.f,0.f,0.f};
  for (int k0 = 0; k0 < Kp; k0 += 32) {
    __builtin_prefetch(arow + k0 + 256, 0, 0);
    HFrag a, b;
    a.q[0] = *(const v8h*)(arow + k0 + grp);
    a.q[1] = *(const v8h*)(arow + k0 + 16 + grp);
    b.q[0] = *(const v8h*)(wrow + k0 + grp);
    b.q[1] = *(const v8h*)(wrow + k0 + 16 + grp);
    acc.v = __builtin_amdgcn_wmma_f32_16x16x32_f16(
        false, a.v, false, b.v, (short)0, acc.v, false, false);
  }

  const int cn = tn + l15;
  if (cn < N) {
    float bv = bias ? bias[cn] : 0.0f;
    float g = 1.0f, bb = 0.0f;
    if (EPI == 1) { g = gamma[cn] * BN_SCALE_C; bb = beta[cn]; }
#pragma unroll
    for (int r = 0; r < 8; ++r) {
      int cm = tm + grp + r;           // D rows: VGPR r -> M=r (+8 for upper lanes)
      if (cm < M) {
        float v = acc.f[r] + bv;
        if (EPI == 1)      { v = v * g + bb; v = v > 0.0f ? v : 0.0f; }
        else if (EPI == 2) { v = 0.5f * v * (1.0f + erff(v * 0.70710678118654752f)); }
        else if (EPI == 3) { v = (v > 20.0f) ? v : log1pf(expf(v)); }
        if (C)  C[(size_t)cm * N + cn] = v;
        if (Ch) Ch[(size_t)cm * N + cn] = (_Float16)v;
      }
    }
  }
}

// ------------- FPS: farthest point sampling (one block per batch) -------------
__global__ void __launch_bounds__(1024)
fps_kernel(const float* __restrict__ xyz, int* __restrict__ fpsIdx,
           float* __restrict__ center) {
  const int b = blockIdx.x;
  const int t = threadIdx.x;          // 0..1023
  __shared__ float sv[1024];
  __shared__ int   si[1024];
  const float* P = xyz + (size_t)b * NN * 3;
  const float px = P[t * 3 + 0], py = P[t * 3 + 1], pz = P[t * 3 + 2];
  float dmin = 1e10f;
  int last = 0;
  if (t == 0) {
    fpsIdx[b * GG] = 0;
    center[(size_t)(b * GG) * 3 + 0] = P[0];
    center[(size_t)(b * GG) * 3 + 1] = P[1];
    center[(size_t)(b * GG) * 3 + 2] = P[2];
  }
  for (int g = 1; g < GG; ++g) {
    float lx = P[last * 3 + 0], ly = P[last * 3 + 1], lz = P[last * 3 + 2];
    float dx = px - lx, dy = py - ly, dz = pz - lz;
    float d = dx * dx + dy * dy + dz * dz;
    dmin = fminf(dmin, d);
    sv[t] = dmin; si[t] = t;
    __syncthreads();
    for (int s = 512; s > 0; s >>= 1) {
      if (t < s) {
        if (sv[t + s] > sv[t] || (sv[t + s] == sv[t] && si[t + s] < si[t])) {
          sv[t] = sv[t + s]; si[t] = si[t + s];
        }
      }
      __syncthreads();
    }
    last = si[0];
    __syncthreads();
    if (t == 0) {
      fpsIdx[b * GG + g] = last;
      center[(size_t)(b * GG + g) * 3 + 0] = P[last * 3 + 0];
      center[(size_t)(b * GG + g) * 3 + 1] = P[last * 3 + 1];
      center[(size_t)(b * GG + g) * 3 + 2] = P[last * 3 + 2];
    }
  }
}

// ------------- KNN (32 nearest of 1024) + neighborhood normalize -------------
__global__ void __launch_bounds__(256)
knn_kernel(const float* __restrict__ xyz, const float* __restrict__ center,
           int* __restrict__ knn, float* __restrict__ nb) {
  const int bg = blockIdx.x;           // 0..511
  const int b = bg >> 7;
  const int t = threadIdx.x;
  __shared__ float d2[NN];
  __shared__ float sv[256];
  __shared__ int   si[256];
  const float* P = xyz + (size_t)b * NN * 3;
  const float cx = center[(size_t)bg * 3 + 0];
  const float cy = center[(size_t)bg * 3 + 1];
  const float cz = center[(size_t)bg * 3 + 2];
  for (int i = t; i < NN; i += 256) {
    float dx = cx - P[i * 3 + 0], dy = cy - P[i * 3 + 1], dz = cz - P[i * 3 + 2];
    d2[i] = dx * dx + dy * dy + dz * dz;
  }
  __syncthreads();
  for (int kk = 0; kk < KK; ++kk) {
    float best = 1e30f; int bi = NN;
    for (int i = t; i < NN; i += 256) {
      float v = d2[i];
      if (v < best || (v == best && i < bi)) { best = v; bi = i; }
    }
    sv[t] = best; si[t] = bi;
    __syncthreads();
    for (int s = 128; s > 0; s >>= 1) {
      if (t < s) {
        if (sv[t + s] < sv[t] || (sv[t + s] == sv[t] && si[t + s] < si[t])) {
          sv[t] = sv[t + s]; si[t] = si[t + s];
        }
      }
      __syncthreads();
    }
    int sel = si[0];
    __syncthreads();
    if (t == 0) {
      knn[bg * KK + kk] = sel;
      d2[sel] = 1e30f;
      size_t r = (size_t)(bg * KK + kk) * 3;
      nb[r + 0] = P[sel * 3 + 0] - cx;
      nb[r + 1] = P[sel * 3 + 1] - cy;
      nb[r + 2] = P[sel * 3 + 2] - cz;
    }
    __syncthreads();
  }
}

// ------------- max over K=32 rows of each group -------------
__global__ void maxpool_kernel(const float* __restrict__ f, float* __restrict__ out, int C) {
  int idx = blockIdx.x * blockDim.x + threadIdx.x;   // over 512*C
  if (idx >= 512 * C) return;
  int gidx = idx / C, c = idx - gidx * C;
  const float* p = f + (size_t)gidx * KK * C + c;
  float m = -1e30f;
  for (int r = 0; r < KK; ++r) m = fmaxf(m, p[(size_t)r * C]);
  out[idx] = m;
}

// ------------- fcat_h = f16([broadcast(fg, K), f2]) -------------
__global__ void concat_kernel(const float* __restrict__ fg, const float* __restrict__ f2,
                              _Float16* __restrict__ fcat) {
  int idx = blockIdx.x * blockDim.x + threadIdx.x;   // 16384*512
  if (idx >= 16384 * 512) return;
  int row = idx >> 9, c = idx & 511;
  int grp = row >> 5;
  float v = (c < 256) ? fg[(size_t)grp * 256 + c]
                      : f2[(size_t)row * 256 + (c - 256)];
  fcat[idx] = (_Float16)v;
}

// ------------- stable argsort of 128 center coords per (b, axis) -------------
__global__ void __launch_bounds__(128)
order_kernel(const float* __restrict__ center, int* __restrict__ perm) {
  const int ax = blockIdx.x % 3;
  const int b = blockIdx.x / 3;
  const int g = threadIdx.x;
  __shared__ float c[GG];
  c[g] = center[(size_t)(b * GG + g) * 3 + ax];
  __syncthreads();
  float v = c[g];
  int rank = 0;
  for (int j = 0; j < GG; ++j) {
    float u = c[j];
    rank += (u < v) || (u == v && j < g);
  }
  perm[(b * 3 + ax) * GG + rank] = g;
}

// ------------- h = gather(tok)+gather(pos), res = 0 -------------
__global__ void build_h_kernel(const float* __restrict__ tok, const float* __restrict__ pos,
                               const int* __restrict__ perm, float* __restrict__ h,
                               float* __restrict__ res) {
  int idx = blockIdx.x * blockDim.x + threadIdx.x;   // B*L*DM
  if (idx >= BB * LL * DM) return;
  int c = idx % DM;
  int tpos = (idx / DM) % LL;
  int b = idx / (DM * LL);
  int g = perm[b * LL + tpos];
  size_t src = (size_t)(b * GG + g) * DM + c;
  h[idx] = tok[src] + pos[src];
  res[idx] = 0.0f;
}

// ------------- block reduction helper (128 threads) -------------
__device__ __forceinline__ float block_sum_128(float v, float* s) {
  int t = threadIdx.x;
  s[t] = v; __syncthreads();
#pragma unroll
  for (int st = 64; st > 0; st >>= 1) {
    if (t < st) s[t] += s[t + st];
    __syncthreads();
  }
  float r = s[0];
  __syncthreads();
  return r;
}

// ------------- res += h ; xh = f16(LN(res)*w+b) -------------
__global__ void __launch_bounds__(128)
resln_kernel(const float* __restrict__ hbuf, float* __restrict__ resbuf,
             _Float16* __restrict__ xh, const float* __restrict__ w,
             const float* __restrict__ b) {
  const int tokI = blockIdx.x;
  const int t = threadIdx.x;
  __shared__ float sred[128];
  const float* hp = hbuf + (size_t)tokI * DM;
  float* rp = resbuf + (size_t)tokI * DM;
  float v0 = hp[t] + rp[t];
  float v1 = hp[t + 128] + rp[t + 128];
  float v2 = hp[t + 256] + rp[t + 256];
  rp[t] = v0; rp[t + 128] = v1; rp[t + 256] = v2;
  float mu = block_sum_128(v0 + v1 + v2, sred) * (1.0f / DM);
  float d0 = v0 - mu, d1 = v1 - mu, d2 = v2 - mu;
  float var = block_sum_128(d0 * d0 + d1 * d1 + d2 * d2, sred) * (1.0f / DM);
  float rs = rsqrtf(var + 1e-5f);
  _Float16* xp = xh + (size_t)tokI * DM;
  xp[t]       = (_Float16)(d0 * rs * w[t]       + b[t]);
  xp[t + 128] = (_Float16)(d1 * rs * w[t + 128] + b[t + 128]);
  xp[t + 256] = (_Float16)(d2 * rs * w[t + 256] + b[t + 256]);
}

// ------------- final: xf = LN(LN(h+res, normf), norm) -------------
__global__ void __launch_bounds__(128)
final_ln_kernel(const float* __restrict__ hbuf, const float* __restrict__ resbuf,
                const float* __restrict__ w1, const float* __restrict__ b1,
                const float* __restrict__ w2, const float* __restrict__ b2,
                float* __restrict__ xf) {
  const int tokI = blockIdx.x;
  const int t = threadIdx.x;
  __shared__ float sred[128];
  const float* hp = hbuf + (size_t)tokI * DM;
  const float* rp = resbuf + (size_t)tokI * DM;
  float v0 = hp[t] + rp[t];
  float v1 = hp[t + 128] + rp[t + 128];
  float v2 = hp[t + 256] + rp[t + 256];
  float mu = block_sum_128(v0 + v1 + v2, sred) * (1.0f / DM);
  float d0 = v0 - mu, d1 = v1 - mu, d2 = v2 - mu;
  float var = block_sum_128(d0 * d0 + d1 * d1 + d2 * d2, sred) * (1.0f / DM);
  float rs = rsqrtf(var + 1e-5f);
  float u0 = d0 * rs * w1[t]       + b1[t];
  float u1 = d1 * rs * w1[t + 128] + b1[t + 128];
  float u2 = d2 * rs * w1[t + 256] + b1[t + 256];
  mu = block_sum_128(u0 + u1 + u2, sred) * (1.0f / DM);
  d0 = u0 - mu; d1 = u1 - mu; d2 = u2 - mu;
  var = block_sum_128(d0 * d0 + d1 * d1 + d2 * d2, sred) * (1.0f / DM);
  rs = rsqrtf(var + 1e-5f);
  float* xp = xf + (size_t)tokI * DM;
  xp[t]       = d0 * rs * w2[t]       + b2[t];
  xp[t + 128] = d1 * rs * w2[t + 128] + b2[t + 128];
  xp[t + 256] = d2 * rs * w2[t + 256] + b2[t + 256];
}

// ------------- causal depthwise conv (D_CONV=4) + SiLU; writes f32 + f16 -------------
__global__ void conv_silu_kernel(const float* __restrict__ xz, const float* __restrict__ cw,
                                 const float* __restrict__ cb, float* __restrict__ ub,
                                 _Float16* __restrict__ ubh) {
  int idx = blockIdx.x * blockDim.x + threadIdx.x;   // B*DI
  if (idx >= BB * DI) return;
  int d = idx % DI, b = idx / DI;
  float w0 = cw[d * 4 + 0], w1 = cw[d * 4 + 1], w2 = cw[d * 4 + 2], w3 = cw[d * 4 + 3];
  float bias = cb[d];
  const float* up = xz + (size_t)b * LL * (2 * DI) + d;
  float* op = ub + (size_t)b * LL * DI + d;
  _Float16* oph = ubh + (size_t)b * LL * DI + d;
  float x0 = 0.f, x1 = 0.f, x2 = 0.f;
  for (int t = 0; t < LL; ++t) {
    float x3 = up[(size_t)t * (2 * DI)];
    float v = x0 * w0 + x1 * w1 + x2 * w2 + x3 * w3 + bias;
    v = v / (1.0f + expf(-v));       // SiLU
    op[(size_t)t * DI] = v;
    oph[(size_t)t * DI] = (_Float16)v;
    x0 = x1; x1 = x2; x2 = x3;
  }
}

// ------------- selective scan + skip + gating; writes f16 y -------------
__global__ void scan_kernel(const float* __restrict__ dt, const float* __restrict__ ub,
                            const float* __restrict__ xdb, const float* __restrict__ xz,
                            const float* __restrict__ A_log, const float* __restrict__ Dp,
                            _Float16* __restrict__ yh) {
  int idx = blockIdx.x * blockDim.x + threadIdx.x;   // B*DI
  if (idx >= BB * DI) return;
  int d = idx % DI, b = idx / DI;
  float a[DS], hs[DS];
#pragma unroll
  for (int s = 0; s < DS; ++s) { a[s] = -expf(A_log[(size_t)d * DS + s]); hs[s] = 0.f; }
  float Dv = Dp[d];
  const float* dtp = dt  + (size_t)b * LL * DI + d;
  const float* up  = ub  + (size_t)b * LL * DI + d;
  const float* zp  = xz  + (size_t)b * LL * (2 * DI) + DI + d;
  const float* bcp = xdb + (size_t)b * LL * (DTR + 2 * DS);
  _Float16* yp = yh + (size_t)b * LL * DI + d;
  for (int t = 0; t < LL; ++t) {
    float dtv = dtp[(size_t)t * DI];
    float uv  = up[(size_t)t * DI];
    const float* row = bcp + (size_t)t * (DTR + 2 * DS);
    float acc = 0.f;
    float du = dtv * uv;
#pragma unroll
    for (int s = 0; s < DS; ++s) {
      float Bv = row[DTR + s];
      float Cv = row[DTR + DS + s];
      hs[s] = expf(dtv * a[s]) * hs[s] + du * Bv;
      acc += hs[s] * Cv;
    }
    float yv = acc + uv * Dv;
    float zv = zp[(size_t)t * (2 * DI)];
    yv *= zv / (1.0f + expf(-zv));   // y * silu(z)
    yp[(size_t)t * DI] = (_Float16)yv;
  }
}

// ------------- mean over L -------------
__global__ void mean_kernel(const float* __restrict__ xf, float* __restrict__ feat) {
  int idx = blockIdx.x * blockDim.x + threadIdx.x;   // B*DM
  if (idx >= BB * DM) return;
  int b = idx / DM, c = idx % DM;
  const float* p = xf + (size_t)b * LL * DM + c;
  float s = 0.f;
  for (int t = 0; t < LL; ++t) s += p[(size_t)t * DM];
  feat[idx] = s * (1.0f / LL);
}

// ------------- host helpers -------------
static inline int pad16(int x) { return (x + 15) & ~15; }
static inline int pad32(int x) { return (x + 31) & ~31; }

static void stage_w(hipStream_t s, const float* W, int N, int Kd, _Float16* dst) {
  int Mp = pad16(N), Kp = pad32(Kd);
  int tot = Mp * Kp;
  cvt_pad_kernel<<<(tot + 255) / 256, 256, 0, s>>>(W, N, Kd, Kd, dst, Mp, Kp);
}
static void stage_a(hipStream_t s, const float* A, int M, int Kd, int lda, _Float16* dst) {
  int Mp = pad16(M), Kp = pad32(Kd);
  int tot = Mp * Kp;
  cvt_pad_kernel<<<(tot + 255) / 256, 256, 0, s>>>(A, M, Kd, lda, dst, Mp, Kp);
}

static void gemm(hipStream_t s, int epi, const _Float16* Ah, const _Float16* Wh,
                 const float* bias, const float* gamma, const float* beta,
                 float* C, _Float16* Ch, int M, int N, int Kp) {
  int tm = pad16(M) / 16, tn = pad16(N) / 16;
  dim3 grid(tm * tn), block(32);
  switch (epi) {
    case 0: gemm_wmma_kernel<0><<<grid, block, 0, s>>>(Ah, Wh, bias, gamma, beta, C, Ch, M, N, Kp, tn); break;
    case 1: gemm_wmma_kernel<1><<<grid, block, 0, s>>>(Ah, Wh, bias, gamma, beta, C, Ch, M, N, Kp, tn); break;
    case 2: gemm_wmma_kernel<2><<<grid, block, 0, s>>>(Ah, Wh, bias, gamma, beta, C, Ch, M, N, Kp, tn); break;
    default: gemm_wmma_kernel<3><<<grid, block, 0, s>>>(Ah, Wh, bias, gamma, beta, C, Ch, M, N, Kp, tn); break;
  }
}

extern "C" void kernel_launch(void* const* d_in, const int* in_sizes, int n_in,
                              void* d_out, int out_size, void* d_ws, size_t ws_size,
                              hipStream_t stream) {
  // ---- inputs (setup_inputs order) ----
  const float* pts        = (const float*)d_in[0];
  const float* enc_w1     = (const float*)d_in[1];
  const float* enc_b1     = (const float*)d_in[2];
  const float* enc_bn1_g  = (const float*)d_in[3];
  const float* enc_bn1_b  = (const float*)d_in[4];
  const float* enc_w2     = (const float*)d_in[5];
  const float* enc_b2     = (const float*)d_in[6];
  const float* enc_w3     = (const float*)d_in[7];
  const float* enc_b3     = (const float*)d_in[8];
  const float* enc_bn2_g  = (const float*)d_in[9];
  const float* enc_bn2_b  = (const float*)d_in[10];
  const float* enc_w4     = (const float*)d_in[11];
  const float* enc_b4     = (const float*)d_in[12];
  const float* pos_w1     = (const float*)d_in[13];
  const float* pos_b1     = (const float*)d_in[14];
  const float* pos_w2     = (const float*)d_in[15];
  const float* pos_b2     = (const float*)d_in[16];
  const float* ln_w       = (const float*)d_in[17];
  const float* ln_b       = (const float*)d_in[18];
  const float* in_proj_w  = (const float*)d_in[19];
  const float* conv_w     = (const float*)d_in[20];
  const float* conv_b     = (const float*)d_in[21];
  const float* x_proj_w   = (const float*)d_in[22];
  const float* dt_proj_w  = (const float*)d_in[23];
  const float* dt_proj_b  = (const float*)d_in[24];
  const float* A_log      = (const float*)d_in[25];
  const float* D_param    = (const float*)d_in[26];
  const float* out_proj_w = (const float*)d_in[27];
  const float* normf_w    = (const float*)d_in[28];
  const float* normf_b    = (const float*)d_in[29];
  const float* norm_w     = (const float*)d_in[30];
  const float* norm_b     = (const float*)d_in[31];
  const float* head_w1    = (const float*)d_in[32];
  const float* head_b1    = (const float*)d_in[33];
  const float* head_bn1_g = (const float*)d_in[34];
  const float* head_bn1_b = (const float*)d_in[35];
  const float* head_w2    = (const float*)d_in[36];
  const float* head_b2    = (const float*)d_in[37];
  const float* head_bn2_g = (const float*)d_in[38];
  const float* head_bn2_b = (const float*)d_in[39];
  const float* head_w3    = (const float*)d_in[40];
  const float* head_b3    = (const float*)d_in[41];
  float* out = (float*)d_out;

  // ---- f32 workspace ----
  float* ws = (float*)d_ws;
  size_t o = 0;
  const size_t BGK = (size_t)BB * GG * KK;       // 16384 rows
  float* nb    = ws + o; o += BGK * 3;
  float* center= ws + o; o += (size_t)BB * GG * 3;
  float* f2    = ws + o; o += BGK * 256;
  float* f4    = ws + o; o += BGK * DM;
  float* fg    = ws + o; o += (size_t)BB * GG * 256;
  float* tok   = ws + o; o += (size_t)BB * GG * DM;
  float* pos   = ws + o; o += (size_t)BB * GG * DM;
  float* hbuf  = ws + o; o += (size_t)BB * LL * DM;
  float* resb  = ws + o; o += (size_t)BB * LL * DM;
  float* xf    = ws + o; o += (size_t)BB * LL * DM;
  float* feat  = ws + o; o += (size_t)BB * DM;
  float* xz    = ws + o; o += (size_t)BB * LL * 2 * DI;
  float* ub    = ws + o; o += (size_t)BB * LL * DI;
  float* xdb   = ws + o; o += (size_t)BB * LL * (DTR + 2 * DS);
  float* dtb   = ws + o; o += (size_t)BB * LL * DI;
  o = (o + 7) & ~(size_t)7;                      // 32B align for f16 arena

  // ---- f16 arena (all buffers padded: rows x16, K x32, 64B-aligned sizes) ----
  _Float16* hbase = (_Float16*)(ws + o);
  size_t ho = 0;
  auto halloc = [&](size_t n) { _Float16* p = hbase + ho; ho += (n + 31) & ~(size_t)31; return p; };
  _Float16* Ah    = halloc(BGK * 32);            // staging for cvt'd A operands (max = nb)
  _Float16* f1h   = halloc(BGK * 128);
  _Float16* fcath = halloc(BGK * 512);
  _Float16* f3h   = halloc(BGK * 512);
  _Float16* pos1h = halloc((size_t)BB * GG * 128);
  _Float16* xh    = halloc((size_t)BB * LL * DM);
  _Float16* ubh   = halloc((size_t)BB * LL * DI);
  _Float16* yh    = halloc((size_t)BB * LL * DI);
  _Float16* h1h   = halloc(16 * 256);
  _Float16* h2h   = halloc(16 * 256);
  // weights (padded f16)
  _Float16* wE1 = halloc(128 * 32);
  _Float16* wE2 = halloc(256 * 128);
  _Float16* wE3 = halloc(512 * 512);
  _Float16* wE4 = halloc((size_t)DM * 512);
  _Float16* wP1 = halloc(128 * 32);
  _Float16* wP2 = halloc((size_t)DM * 128);
  _Float16* wIp[DEPTH]; _Float16* wXp[DEPTH]; _Float16* wDt[DEPTH]; _Float16* wOp[DEPTH];
  for (int l = 0; l < DEPTH; ++l) {
    wIp[l] = halloc((size_t)2 * DI * DM);        // 1536 x 384
    wXp[l] = halloc((size_t)64 * DI);            // pad16(56) x 768
    wDt[l] = halloc((size_t)DI * 32);            // 768 x pad32(24)
    wOp[l] = halloc((size_t)DM * DI);            // 384 x 768
  }
  _Float16* wH1 = halloc((size_t)256 * DM);
  _Float16* wH2 = halloc((size_t)256 * 256);
  _Float16* wH3 = halloc((size_t)48 * 256);      // pad16(40) x 256
  // int scratch after f16 arena
  int* ip = (int*)(hbase + ho);
  int* fpsIdx = ip;
  int* knnIdx = ip + 512;
  int* perm   = ip + 512 + 16384;

  // ---- stage all weights to padded f16 ----
  stage_w(stream, enc_w1, 128, 3, wE1);
  stage_w(stream, enc_w2, 256, 128, wE2);
  stage_w(stream, enc_w3, 512, 512, wE3);
  stage_w(stream, enc_w4, DM, 512, wE4);
  stage_w(stream, pos_w1, 128, 3, wP1);
  stage_w(stream, pos_w2, DM, 128, wP2);
  for (int l = 0; l < DEPTH; ++l) {
    stage_w(stream, in_proj_w  + (size_t)l * 2 * DI * DM, 2 * DI, DM, wIp[l]);
    stage_w(stream, x_proj_w   + (size_t)l * (DTR + 2 * DS) * DI, DTR + 2 * DS, DI, wXp[l]);
    stage_w(stream, dt_proj_w  + (size_t)l * DI * DTR, DI, DTR, wDt[l]);
    stage_w(stream, out_proj_w + (size_t)l * DM * DI, DM, DI, wOp[l]);
  }
  stage_w(stream, head_w1, 256, DM, wH1);
  stage_w(stream, head_w2, 256, 256, wH2);
  stage_w(stream, head_w3, CLS, 256, wH3);

  // ---- grouping ----
  fps_kernel<<<BB, 1024, 0, stream>>>(pts, fpsIdx, center);
  knn_kernel<<<BB * GG, 256, 0, stream>>>(pts, center, knnIdx, nb);

  // ---- encoder ----
  stage_a(stream, nb, (int)BGK, 3, 3, Ah);
  gemm(stream, 1, Ah, wE1, enc_b1, enc_bn1_g, enc_bn1_b, nullptr, f1h, (int)BGK, 128, 32);
  gemm(stream, 0, f1h, wE2, enc_b2, nullptr, nullptr, f2, nullptr, (int)BGK, 256, 128);
  {
    int tot = 512 * 256;
    maxpool_kernel<<<(tot + 255) / 256, 256, 0, stream>>>(f2, fg, 256);
  }
  {
    int tot = (int)(BGK * 512);
    concat_kernel<<<(tot + 255) / 256, 256, 0, stream>>>(fg, f2, fcath);
  }
  gemm(stream, 1, fcath, wE3, enc_b3, enc_bn2_g, enc_bn2_b, nullptr, f3h, (int)BGK, 512, 512);
  gemm(stream, 0, f3h, wE4, enc_b4, nullptr, nullptr, f4, nullptr, (int)BGK, DM, 512);
  {
    int tot = 512 * DM;
    maxpool_kernel<<<(tot + 255) / 256, 256, 0, stream>>>(f4, tok, DM);
  }

  // ---- positional embedding ----
  stage_a(stream, center, BB * GG, 3, 3, Ah);
  gemm(stream, 2, Ah, wP1, pos_b1, nullptr, nullptr, nullptr, pos1h, BB * GG, 128, 32);
  gemm(stream, 0, pos1h, wP2, pos_b2, nullptr, nullptr, pos, nullptr, BB * GG, DM, 128);

  // ---- token reordering + h/res init ----
  order_kernel<<<BB * 3, GG, 0, stream>>>(center, perm);
  {
    int tot = BB * LL * DM;
    build_h_kernel<<<(tot + 255) / 256, 256, 0, stream>>>(tok, pos, perm, hbuf, resb);
  }

  // ---- 12 Mamba layers ----
  const int NT = BB * LL;          // 1536 tokens
  for (int l = 0; l < DEPTH; ++l) {
    const float* lnw = ln_w + (size_t)l * DM;
    const float* lnb = ln_b + (size_t)l * DM;
    const float* cw  = conv_w + (size_t)l * DI * 4;
    const float* cb  = conv_b + (size_t)l * DI;
    const float* dpb = dt_proj_b + (size_t)l * DI;
    const float* Al  = A_log + (size_t)l * DI * DS;
    const float* Dp  = D_param + (size_t)l * DI;

    resln_kernel<<<NT, 128, 0, stream>>>(hbuf, resb, xh, lnw, lnb);
    gemm(stream, 0, xh, wIp[l], nullptr, nullptr, nullptr, xz, nullptr, NT, 2 * DI, DM);
    conv_silu_kernel<<<(BB * DI + 255) / 256, 256, 0, stream>>>(xz, cw, cb, ub, ubh);
    gemm(stream, 0, ubh, wXp[l], nullptr, nullptr, nullptr, xdb, nullptr, NT, DTR + 2 * DS, DI);
    stage_a(stream, xdb, NT, DTR, DTR + 2 * DS, Ah);
    gemm(stream, 3, Ah, wDt[l], dpb, nullptr, nullptr, dtb, nullptr, NT, DI, 32);
    scan_kernel<<<(BB * DI + 255) / 256, 256, 0, stream>>>(dtb, ub, xdb, xz, Al, Dp, yh);
    gemm(stream, 0, yh, wOp[l], nullptr, nullptr, nullptr, hbuf, nullptr, NT, DM, DI);
  }

  // ---- final norms, pooling, head ----
  final_ln_kernel<<<NT, 128, 0, stream>>>(hbuf, resb, normf_w, normf_b, norm_w, norm_b, xf);
  mean_kernel<<<(BB * DM + 255) / 256, 256, 0, stream>>>(xf, feat);
  stage_a(stream, feat, BB, DM, DM, Ah);
  gemm(stream, 1, Ah, wH1, head_b1, head_bn1_g, head_bn1_b, nullptr, h1h, BB, 256, DM);
  gemm(stream, 1, h1h, wH2, head_b2, head_bn2_g, head_bn2_b, nullptr, h2h, BB, 256, 256);
  gemm(stream, 0, h2h, wH3, head_b3, nullptr, nullptr, out, nullptr, BB, CLS, 256);

  (void)in_sizes; (void)n_in; (void)out_size; (void)ws_size; (void)fpsIdx; (void)knnIdx;
}
